// RGCNConv_4398046511496
// MI455X (gfx1250) — matrix operand order repaired and checked by reading, not compiled, not run on hardware
//
#include <hip/hip_runtime.h>

#define D 128

typedef __attribute__((ext_vector_type(16))) unsigned short v16u;
typedef __attribute__((ext_vector_type(16))) __bf16        v16bf;
typedef __attribute__((ext_vector_type(8)))  float         v8f;
typedef __attribute__((ext_vector_type(4)))  unsigned int  u32x4;
typedef __attribute__((ext_vector_type(8)))  int           i32x8;
typedef __attribute__((ext_vector_type(4)))  int           i32x4;

__device__ __forceinline__ unsigned short f32_bf16(float f) {
    unsigned u = __float_as_uint(f);
    u += 0x7FFFu + ((u >> 16) & 1u);   // round-to-nearest-even
    return (unsigned short)(u >> 16);
}

// ---------------------------------------------------------------------------
// One-time weight prep: W (128x128 f32, row-major [k][n]) -> Wt (bf16, [n][k])
// ---------------------------------------------------------------------------
__global__ void convert_weight(const float* __restrict__ W,
                               unsigned short* __restrict__ Wt) {
    int idx = blockIdx.x * blockDim.x + threadIdx.x;   // 16384 threads total
    int k = idx >> 7, n = idx & 127;
    Wt[n * D + k] = f32_bf16(W[idx]);
}

// ---------------------------------------------------------------------------
// Edge scatter: agg[dst[e]] += x[src[e]] (128 floats), cnt[dst[e]] += 1.
// One edge per 32 lanes; each lane handles 4 contiguous floats.
// ---------------------------------------------------------------------------
__global__ void scatter_mean_accum(const float* __restrict__ x,
                                   const int* __restrict__ src,
                                   const int* __restrict__ dst,
                                   float* __restrict__ agg,
                                   float* __restrict__ cnt, int E) {
    long long t = (long long)blockIdx.x * blockDim.x + threadIdx.x;
    long long e = t >> 5;
    if (e >= E) return;
    int lane = (int)(t & 31);
    int s = src[e];
    int d = dst[e];
    if (e + 8 < E) __builtin_prefetch(x + (long long)src[e + 8] * D + lane * 4, 0, 1);
    const float4 v = *(const float4*)(x + (long long)s * D + lane * 4);
    float* ap = agg + (long long)d * D + lane * 4;
    atomicAdd(ap + 0, v.x);
    atomicAdd(ap + 1, v.y);
    atomicAdd(ap + 2, v.z);
    atomicAdd(ap + 3, v.w);
    if (lane == 0) atomicAdd(cnt + d, 1.0f);
}

// agg[row] *= 1/max(cnt[row],1)
__global__ void normalize_rows(float* __restrict__ agg,
                               const float* __restrict__ cnt, int n) {
    long long t = (long long)blockIdx.x * blockDim.x + threadIdx.x;
    long long row = t >> 5;
    if (row >= n) return;
    int c4 = (int)(t & 31);
    float s = 1.0f / fmaxf(cnt[row], 1.0f);
    float4* p = (float4*)(agg + row * D + c4 * 4);
    float4 v = *p;
    v.x *= s; v.y *= s; v.z *= s; v.w *= s;
    *p = v;
}

// ---------------------------------------------------------------------------
// GEMM: Out[(NT*16)xD slab] = X @ W (+bias)  or  Out += X @ W (accumulate=1)
// Block = 256 threads = 8 wave32. Weight (pre-converted bf16 [n][k], 32 KB)
// pulled into LDS by the Tensor Data Mover; A slab converted f32->bf16 into
// LDS. Each wave owns one 16-wide N tile and NT M tiles, reusing its B
// fragment across M: NT*4 v_wmma_f32_16x16x32_bf16 per wave. NT is a
// compile-time constant so accumulators stay in fixed VGPRs (no v_movrel).
// ---------------------------------------------------------------------------
template <int NT>
__global__ void __launch_bounds__(256)
gemm_bf16_wmma(const float* __restrict__ X,
               const unsigned short* __restrict__ Wt,   // bf16 [n][k]
               const float* __restrict__ bias, float* __restrict__ Out,
               int accumulate) {
    __shared__ unsigned short sA[NT * 16 * D];   // [m][k]
    __shared__ unsigned short sWt[D * D];        // [n][k], 32 KB

    const int t = threadIdx.x;
    const long long r0 = (long long)blockIdx.x * (NT * 16);
    const float* Xb = X + r0 * D;

    // --- TDM: async copy 32 KB weight image (bf16) global -> LDS -----------
    if (t < 32) {  // wave 0 only; TDM ignores EXEC, issue exactly once per block
        unsigned long long ga = (unsigned long long)(uintptr_t)Wt;
        unsigned ldsOff = (unsigned)(uintptr_t)(&sWt[0]);
        u32x4 g0;
        g0[0] = 1u;                                    // count=1 valid descriptor
        g0[1] = ldsOff;                                // lds_addr
        g0[2] = (unsigned)(ga & 0xFFFFFFFFu);          // global_addr[31:0]
        g0[3] = (unsigned)((ga >> 32) & 0x01FFFFFFu)   // global_addr[56:32]
              | (2u << 30);                            // type=2 ("image")
        i32x8 g1;
        g1[0] = (1 << 16);                             // data_size=1 -> 2-byte elems
        g1[1] = (int)(16384u << 16);                   // tensor_dim0 = 16384 (lo16)
        g1[2] = (1 << 16);                             // tensor_dim0 hi=0, tensor_dim1=1
        g1[3] = (int)(16384u << 16);                   // tile_dim0 = 16384
        g1[4] = 1;                                     // tile_dim1 = 1
        g1[5] = 16384;                                 // tensor_dim0_stride
        g1[6] = 0;
        g1[7] = 0;
        i32x4 gz4 = {0, 0, 0, 0};
        i32x8 gz8 = {0, 0, 0, 0, 0, 0, 0, 0};
        __builtin_amdgcn_tensor_load_to_lds(g0, g1, gz4, gz4, gz8, 0);
        __builtin_amdgcn_s_wait_tensorcnt(0);
    }

    // --- Stage A slab (NT*16 x 128), f32 -> bf16 ---------------------------
#pragma unroll
    for (int j = 0; j < NT * 8; ++j) {
        int idx = j * 256 + t;                          // NT*2048 total
        sA[idx] = f32_bf16(Xb[idx]);
    }
    __syncthreads();

    const int wave = t >> 5;        // 0..7 -> N tile
    const int lane = t & 31;
    const int lh   = lane >> 4;     // lane-half (ISA fragment layout)
    const int ln   = lane & 15;
    const int n    = wave * 16 + ln;

    // C/D layout: VGPR i -> row (i + 8*lh), col n
    v8f acc[NT];
    if (accumulate) {
#pragma unroll
        for (int m = 0; m < NT; ++m) {
            const float* ob = Out + (r0 + m * 16) * D;
#pragma unroll
            for (int i = 0; i < 8; ++i) acc[m][i] = ob[(i + 8 * lh) * D + n];
        }
    } else {
        float b = bias[n];
#pragma unroll
        for (int m = 0; m < NT; ++m)
#pragma unroll
            for (int i = 0; i < 8; ++i) acc[m][i] = b;
    }

#pragma unroll
    for (int kt = 0; kt < 4; ++kt) {
        const int kb = kt * 32;
        // B 32x16 bf16 layout: col = ln; halves 0..7 -> K=kb+8*lh+i, halves 8..15 -> K=kb+16+8*lh+i
        v16u b;
        const unsigned short* br = sWt + n * D + kb + 8 * lh;
#pragma unroll
        for (int i = 0; i < 8; ++i) b[i] = br[i];
#pragma unroll
        for (int i = 0; i < 8; ++i) b[8 + i] = br[16 + i];

#pragma unroll
        for (int m = 0; m < NT; ++m) {
            // A 16x32 bf16 layout: row = ln; halves 0..7 -> K=kb+8*lh+i,
            //                                halves 8..15 -> K=kb+16+8*lh+i
            v16u a;
            const unsigned short* ar = sA + (m * 16 + ln) * D + kb + 8 * lh;
#pragma unroll
            for (int i = 0; i < 8; ++i) a[i]     = ar[i];
#pragma unroll
            for (int i = 0; i < 8; ++i) a[8 + i] = ar[16 + i];
            acc[m] = __builtin_amdgcn_wmma_f32_16x16x32_bf16(
                false, __builtin_bit_cast(v16bf, a),
                false, __builtin_bit_cast(v16bf, b),
                (short)0, acc[m], false, false);
        }
    }

#pragma unroll
    for (int m = 0; m < NT; ++m) {
        float* ob = Out + (r0 + m * 16) * D;
#pragma unroll
        for (int i = 0; i < 8; ++i) ob[(i + 8 * lh) * D + n] = acc[m][i];
    }
}

// ---------------------------------------------------------------------------
extern "C" void kernel_launch(void* const* d_in, const int* in_sizes, int n_in,
                              void* d_out, int out_size, void* d_ws, size_t ws_size,
                              hipStream_t stream) {
    const int NA = 100000, NF = 30000, NI = 8000, NP = 200000;

    const float* x_author = (const float*)d_in[0];
    const float* x_fos    = (const float*)d_in[1];
    const float* x_inst   = (const float*)d_in[2];
    const float* x_paper  = (const float*)d_in[3];

    const int* srcp[7]; const int* dstp[7]; int E[7];
    for (int r = 0; r < 7; ++r) {
        srcp[r] = (const int*)d_in[4 + 2 * r];
        dstp[r] = (const int*)d_in[5 + 2 * r];
        E[r]    = in_sizes[4 + 2 * r];
    }
    const float* Wroot[4]; const float* broot[4];
    for (int i = 0; i < 4; ++i) {
        Wroot[i] = (const float*)d_in[18 + 2 * i];
        broot[i] = (const float*)d_in[19 + 2 * i];
    }
    const float* Wrel[7];
    for (int r = 0; r < 7; ++r) Wrel[r] = (const float*)d_in[26 + r];

    float* out        = (float*)d_out;
    float* out_author = out;
    float* out_fos    = out + (long long)NA * D;
    float* out_inst   = out + (long long)(NA + NF) * D;
    float* out_paper  = out + (long long)(NA + NF + NI) * D;

    // Workspace layout: agg (200000*128 f32) | cnt (200000 f32) | 11x bf16 Wt
    float* agg = (float*)d_ws;
    float* cnt = (float*)((char*)d_ws + (size_t)NP * D * 4);
    unsigned short* wtBase =
        (unsigned short*)((char*)d_ws + (size_t)NP * D * 4 + (size_t)NP * 4);
    unsigned short* wt[11];
    for (int i = 0; i < 11; ++i) wt[i] = wtBase + (size_t)i * D * D;

    // One-time weight conversion (f32 [k][n] -> bf16 [n][k])
    for (int i = 0; i < 4; ++i)
        convert_weight<<<64, 256, 0, stream>>>(Wroot[i], wt[i]);
    for (int r = 0; r < 7; ++r)
        convert_weight<<<64, 256, 0, stream>>>(Wrel[r], wt[4 + r]);

    // Full 64-row blocks + 16-row-aligned tail (all row counts are %16 == 0)
    auto launch_gemm = [&](const float* X, const unsigned short* Wtb,
                           const float* bias, float* Out, int rows, int accum) {
        int full = rows / 64;
        if (full > 0)
            gemm_bf16_wmma<4><<<full, 256, 0, stream>>>(X, Wtb, bias, Out, accum);
        int remT = (rows - full * 64) / 16;
        if (remT > 0)
            gemm_bf16_wmma<1><<<remT, 256, 0, stream>>>(
                X + (long long)full * 64 * D, Wtb, bias,
                Out + (long long)full * 64 * D, accum);
    };

    // Root transforms: fully initialize d_out
    launch_gemm(x_author, wt[0], broot[0], out_author, NA, 0);
    launch_gemm(x_fos,    wt[1], broot[1], out_fos,    NF, 0);
    launch_gemm(x_inst,   wt[2], broot[2], out_inst,   NI, 0);
    launch_gemm(x_paper,  wt[3], broot[3], out_paper,  NP, 0);

    // relation -> (source features, n_dst, output section)
    const float* xs[7]  = { x_author, x_inst, x_author, x_paper, x_paper, x_paper, x_fos };
    const int   ndst[7] = { NI,       NA,     NP,       NA,      NP,      NF,      NP    };
    float*      osec[7] = { out_inst, out_author, out_paper, out_author, out_paper, out_fos, out_paper };

    for (int r = 0; r < 7; ++r) {
        (void)hipMemsetAsync(agg, 0, (size_t)ndst[r] * D * sizeof(float), stream);
        (void)hipMemsetAsync(cnt, 0, (size_t)ndst[r] * sizeof(float), stream);

        long long nThr = (long long)E[r] * 32;
        int sBlocks = (int)((nThr + 255) / 256);
        scatter_mean_accum<<<sBlocks, 256, 0, stream>>>(xs[r], srcp[r], dstp[r], agg, cnt, E[r]);

        long long nThrN = (long long)ndst[r] * 32;
        int nBlocks = (int)((nThrN + 255) / 256);
        normalize_rows<<<nBlocks, 256, 0, stream>>>(agg, cnt, ndst[r]);

        launch_gemm(agg, wt[4 + r], nullptr, osec[r], ndst[r], 1);
    }
}
